// Synth_Attention_54219667145484
// MI455X (gfx1250) — compile-verified, matrix operand order checked by reading, not run on hardware
//
#include <hip/hip_runtime.h>
#include <hip/hip_bf16.h>

// ---------- problem constants ----------
#define HH   8
#define DD   256
#define HDIM 32
#define BB   16
#define QQ   512
#define S1L  64
#define NT   448   // task rows
#define NS   64    // station rows

typedef __attribute__((ext_vector_type(16))) _Float16 v16h;
typedef __attribute__((ext_vector_type(8)))  float    v8f;

// ---------------------------------------------------------------------------
// WMMA helper: D = A(16x32 f16) * B(32x16 f16) + C(f32)
// ---------------------------------------------------------------------------
__device__ __forceinline__ v8f wmma_fma(v16h a, v16h b, v8f c) {
  return __builtin_amdgcn_wmma_f32_16x16x32_f16(
      /*neg_a=*/false, a, /*neg_b=*/false, b,
      /*c_mod=*/(short)0, c, /*reuse_a=*/false, /*reuse_b=*/false);
}

// A-operand gather (16-bit A 16x32 layout, ISA 7.12.2):
// lane = half*16 + row ; VGPR v holds K pair {kb, kb+1},
// kb = (v>>2)*16 + half*8 + (v&3)*2.  Source is row-major [row][k].
__device__ __forceinline__ v16h ldA_f16(const _Float16* base, int ld, int lane) {
  const int row = lane & 15, half = lane >> 4;
  v16h a;
#pragma unroll
  for (int v = 0; v < 8; ++v) {
    const int kb = ((v >> 2) << 4) | (half << 3) | ((v & 3) << 1);
    a[2 * v]     = base[row * ld + kb];
    a[2 * v + 1] = base[row * ld + kb + 1];
  }
  return a;
}

__device__ __forceinline__ v16h ldA_f32(const float* base, int ld, int lane) {
  const int row = lane & 15, half = lane >> 4;
  v16h a;
#pragma unroll
  for (int v = 0; v < 8; ++v) {
    const int kb = ((v >> 2) << 4) | (half << 3) | ((v & 3) << 1);
    a[2 * v]     = (_Float16)base[row * ld + kb];
    a[2 * v + 1] = (_Float16)base[row * ld + kb + 1];
  }
  return a;
}

// B-operand gather from a row-major [k][n] source (B is 32x16, K x N):
// lane = half*16 + col ; VGPR v holds K pair {kb, kb+1} for column col.
__device__ __forceinline__ v16h ldB_rows_f16(const _Float16* base, int ld, int lane) {
  const int col = lane & 15, half = lane >> 4;
  v16h b;
#pragma unroll
  for (int v = 0; v < 8; ++v) {
    const int kb = ((v >> 2) << 4) | (half << 3) | ((v & 3) << 1);
    b[2 * v]     = base[kb * ld + col];
    b[2 * v + 1] = base[(kb + 1) * ld + col];
  }
  return b;
}

__device__ __forceinline__ v16h ldB_rows_f32(const float* base, int ld, int lane) {
  const int col = lane & 15, half = lane >> 4;
  v16h b;
#pragma unroll
  for (int v = 0; v < 8; ++v) {
    const int kb = ((v >> 2) << 4) | (half << 3) | ((v & 3) << 1);
    b[2 * v]     = (_Float16)base[kb * ld + col];
    b[2 * v + 1] = (_Float16)base[(kb + 1) * ld + col];
  }
  return b;
}

// ---------------------------------------------------------------------------
// Kernel 1: the 7 head projections, f32 inputs -> f16 workspace.
// One wave (32 threads) computes one 16x32 output tile with 16 WMMAs.
// ---------------------------------------------------------------------------
__global__ __launch_bounds__(32) void proj_kernel(
    const float* __restrict__ h_fea,
    const float* __restrict__ Wq_custom1, const float* __restrict__ Wq_custom,
    const float* __restrict__ Wk_custom,  const float* __restrict__ Wv_custom,
    const float* __restrict__ Wq_charge1, const float* __restrict__ Wk_charge,
    const float* __restrict__ Wv_charge,
    _Float16* __restrict__ Qt1, _Float16* __restrict__ Qt,
    _Float16* __restrict__ Kc,  _Float16* __restrict__ Vc,
    _Float16* __restrict__ Qs,  _Float16* __restrict__ Ks,
    _Float16* __restrict__ Vs) {
  const int lane = threadIdx.x;
  int id = blockIdx.x;
  const int TASK_BLOCKS = 4 * HH * BB * 28;

  const float* W;
  _Float16*    Out;
  int h, b, t, rows, row0;

  if (id < TASK_BLOCKS) {
    const int proj = id / (HH * BB * 28);
    int rem = id % (HH * BB * 28);
    h = rem / (BB * 28); rem %= (BB * 28);
    b = rem / 28; t = rem % 28;
    row0 = S1L + t * 16;
    rows = NT;
    if      (proj == 0) { W = Wq_custom1; Out = Qt1; }
    else if (proj == 1) { W = Wq_custom;  Out = Qt;  }
    else if (proj == 2) { W = Wk_custom;  Out = Kc;  }
    else                { W = Wv_custom;  Out = Vc;  }
  } else {
    id -= TASK_BLOCKS;
    const int proj = id / (HH * BB * 4);
    int rem = id % (HH * BB * 4);
    h = rem / (BB * 4); rem %= (BB * 4);
    b = rem / 4; t = rem % 4;
    row0 = t * 16;
    rows = NS;
    if      (proj == 0) { W = Wq_charge1; Out = Qs; }
    else if (proj == 1) { W = Wk_charge;  Out = Ks; }
    else                { W = Wv_charge;  Out = Vs; }
  }

  const float* X  = h_fea + ((size_t)b * QQ + row0) * DD;
  const float* Wh = W + (size_t)h * DD * HDIM;

  v8f c0 = {}, c1 = {};
#pragma unroll
  for (int k0 = 0; k0 < DD; k0 += 32) {
    v16h a  = ldA_f32(X + k0, DD, lane);
    v16h b0 = ldB_rows_f32(Wh + (size_t)k0 * HDIM,      HDIM, lane);
    v16h b1 = ldB_rows_f32(Wh + (size_t)k0 * HDIM + 16, HDIM, lane);
    c0 = wmma_fma(a, b0, c0);
    c1 = wmma_fma(a, b1, c1);
  }

  const int col = lane & 15, half = lane >> 4;
  _Float16* obase = Out + ((size_t)(h * BB + b) * rows + t * 16) * HDIM;
#pragma unroll
  for (int r = 0; r < 8; ++r) {
    const int rr = half * 8 + r;
    obase[(size_t)rr * HDIM + col]      = (_Float16)c0[r];
    obase[(size_t)rr * HDIM + col + 16] = (_Float16)c1[r];
  }
}

// ---------------------------------------------------------------------------
// Kernel 2/3: attention with cross-head fuse-MLP. One block = one (b, 16-row
// tile), 8 waves = 8 heads. All raw scores for the tile stay LDS-resident
// (up to 256 KB of CDNA5's 320 KB WGP LDS), so aux is read exactly once.
// TARGE=true  : task rows, both attends (M=448 vs Kc/Vc + M=64 vs Ks/Vs)
// TARGE=false : station rows, one attend (M=448 vs Kc/Vc)
// ---------------------------------------------------------------------------
template <bool TARGE>
__global__ __launch_bounds__(256) void attn_kernel(
    const _Float16* __restrict__ Qt1, const _Float16* __restrict__ Qt,
    const _Float16* __restrict__ Kc,  const _Float16* __restrict__ Vc,
    const _Float16* __restrict__ Qs,  const _Float16* __restrict__ Ks,
    const _Float16* __restrict__ Vs,
    const float* __restrict__ aux,
    const float* __restrict__ W1, const float* __restrict__ b1,
    const float* __restrict__ W2, const float* __restrict__ b2,
    float* __restrict__ heads) {
  constexpr int MTOT = TARGE ? (NT + NS) : NT;  // 512 or 448 score columns
  const int nt  = TARGE ? 28 : 4;
  const int b   = blockIdx.x / nt;
  const int t   = blockIdx.x % nt;
  const int n0  = t * 16;
  const int qn0 = TARGE ? (S1L + n0) : n0;

  extern __shared__ float smem[];
  float* Ssc = smem;                  // [8][16][MTOT] raw->fused->softmaxed
  float* Wc  = smem + 8 * 16 * MTOT;  // 408 floats of MLP weights

  const int tid = threadIdx.x;
  for (int i = tid; i < 408; i += 256) {
    float v;
    if      (i < 256) v = W1[i];
    else if (i < 272) v = b1[i - 256];
    else if (i < 400) v = W2[i - 272];
    else              v = b2[i - 400];
    Wc[i] = v;
  }

  const int wave = tid >> 5, lane = tid & 31;
  const int h = wave;
  const int col = lane & 15, half = lane >> 4;

  // ---- Phase 1: raw scores S = Q K^T into LDS -------------------------------
  {
    const _Float16* Qsrc =
        TARGE ? (Qt1 + ((size_t)(h * BB + b) * NT + n0) * HDIM)
              : (Qs  + ((size_t)(h * BB + b) * NS + n0) * HDIM);
    v16h aq = ldA_f16(Qsrc, HDIM, lane);

    for (int mt = 0; mt < 28; ++mt) {
      const _Float16* Ktile = Kc + ((size_t)(h * BB + b) * NT + mt * 16) * HDIM;
      if (mt + 1 < 28)
        __builtin_prefetch(Ktile + 16 * HDIM, 0, 1);
      // Row-major K tile loaded with the A-gather == B = K^T operand.
      v16h bk = ldA_f16(Ktile, HDIM, lane);
      v8f c = {};
      c = wmma_fma(aq, bk, c);
      float* sdst = Ssc + (size_t)(h * 16) * MTOT + mt * 16 + col;
#pragma unroll
      for (int r = 0; r < 8; ++r) sdst[(size_t)(half * 8 + r) * MTOT] = c[r];
    }

    if (TARGE) {
      const _Float16* Q2 = Qt + ((size_t)(h * BB + b) * NT + n0) * HDIM;
      v16h aq2 = ldA_f16(Q2, HDIM, lane);
      for (int mt = 0; mt < 4; ++mt) {
        const _Float16* Ktile = Ks + ((size_t)(h * BB + b) * NS + mt * 16) * HDIM;
        v16h bk = ldA_f16(Ktile, HDIM, lane);
        v8f c = {};
        c = wmma_fma(aq2, bk, c);
        float* sdst = Ssc + (size_t)(h * 16) * MTOT + NT + mt * 16 + col;
#pragma unroll
        for (int r = 0; r < 8; ++r) sdst[(size_t)(half * 8 + r) * MTOT] = c[r];
      }
    }
  }
  __syncthreads();

  // ---- Phase 2: cross-head fuse MLP, in place -------------------------------
  {
    const int npos = 16 * MTOT;
    for (int pos = tid; pos < npos; pos += 256) {
      const int n = pos / MTOT, m = pos - n * MTOT;
      float x[16];
#pragma unroll
      for (int hh = 0; hh < 8; ++hh) x[hh] = Ssc[(size_t)(hh * 16 + n) * MTOT + m];
      int qm;
      if (TARGE) qm = (m < NT) ? (S1L + m) : (m - NT);
      else       qm = S1L + m;
#pragma unroll
      for (int hh = 0; hh < 8; ++hh)
        x[8 + hh] = aux[(((size_t)hh * BB + b) * QQ + (qn0 + n)) * QQ + qm];
      float y[16];
#pragma unroll
      for (int j = 0; j < 16; ++j) {
        float acc = Wc[256 + j];
#pragma unroll
        for (int i = 0; i < 16; ++i) acc = fmaf(x[i], Wc[i * 16 + j], acc);
        y[j] = fmaxf(acc, 0.f);
      }
#pragma unroll
      for (int hh = 0; hh < 8; ++hh) {
        float acc = Wc[400 + hh];
#pragma unroll
        for (int j = 0; j < 16; ++j) acc = fmaf(y[j], Wc[272 + j * 8 + hh], acc);
        Ssc[(size_t)(hh * 16 + n) * MTOT + m] = acc;
      }
    }
  }
  __syncthreads();

  // ---- Phase 3: per-(head,row,segment) softmax ------------------------------
  {
    if (TARGE) {  // 8 heads * 16 rows * 2 segments == 256 threads
      const int hh = tid >> 5;
      const int rw = tid & 31;
      const int row = rw >> 1, seg = rw & 1;
      float* p = Ssc + (size_t)(hh * 16 + row) * MTOT;
      const int ms = seg ? NT : 0, me = seg ? MTOT : NT;
      float mx = -3.4e38f;
      for (int m = ms; m < me; ++m) mx = fmaxf(mx, p[m]);
      float sum = 0.f;
      for (int m = ms; m < me; ++m) { float e = __expf(p[m] - mx); p[m] = e; sum += e; }
      const float inv = 1.f / sum;
      for (int m = ms; m < me; ++m) p[m] *= inv;
    } else {      // 8 heads * 16 rows == 128 threads
      if (tid < 128) {
        const int hh = tid >> 4, row = tid & 15;
        float* p = Ssc + (size_t)(hh * 16 + row) * MTOT;
        float mx = -3.4e38f;
        for (int m = 0; m < NT; ++m) mx = fmaxf(mx, p[m]);
        float sum = 0.f;
        for (int m = 0; m < NT; ++m) { float e = __expf(p[m] - mx); p[m] = e; sum += e; }
        const float inv = 1.f / sum;
        for (int m = 0; m < NT; ++m) p[m] *= inv;
      }
    }
  }
  __syncthreads();

  // ---- Phase 4: O = P @ V (sum of both attends for TARGE) -------------------
  {
    v8f o0 = {}, o1 = {};
    const float* Prow = Ssc + (size_t)(h * 16) * MTOT;
#pragma unroll
    for (int st = 0; st < 14; ++st) {
      const int m0 = st * 32;
      v16h ap = ldA_f32(Prow + m0, MTOT, lane);  // probabilities f32 -> f16
      const _Float16* Vt = Vc + ((size_t)(h * BB + b) * NT + m0) * HDIM;
      v16h bv0 = ldB_rows_f16(Vt,      HDIM, lane);
      v16h bv1 = ldB_rows_f16(Vt + 16, HDIM, lane);
      o0 = wmma_fma(ap, bv0, o0);
      o1 = wmma_fma(ap, bv1, o1);
    }
    if (TARGE) {
#pragma unroll
      for (int st = 0; st < 2; ++st) {
        const int m0 = NT + st * 32;
        v16h ap = ldA_f32(Prow + m0, MTOT, lane);
        const _Float16* Vt = Vs + ((size_t)(h * BB + b) * NS + st * 32) * HDIM;
        v16h bv0 = ldB_rows_f16(Vt,      HDIM, lane);
        v16h bv1 = ldB_rows_f16(Vt + 16, HDIM, lane);
        o0 = wmma_fma(ap, bv0, o0);
        o1 = wmma_fma(ap, bv1, o1);
      }
    }
    const int qbase = TARGE ? (S1L + n0) : n0;
    float* hd = heads + ((size_t)(h * BB + b) * QQ + qbase) * HDIM;
#pragma unroll
    for (int r = 0; r < 8; ++r) {
      const int rr = half * 8 + r;
      hd[(size_t)rr * HDIM + col]      = o0[r];
      hd[(size_t)rr * HDIM + col + 16] = o1[r];
    }
  }
}

// ---------------------------------------------------------------------------
// Kernel 4: out[b,q,:] = concat_h heads[h,b,q,:] @ W_out  (8192x256x256 GEMM)
// 8 waves/block, one 16x16 output tile per wave, 8 chained WMMAs (K=256).
// ---------------------------------------------------------------------------
__global__ __launch_bounds__(256) void outproj_kernel(
    const float* __restrict__ heads, const float* __restrict__ Wout,
    float* __restrict__ out) {
  const int wave = threadIdx.x >> 5, lane = threadIdx.x & 31;
  const int tile  = blockIdx.x * 8 + wave;  // 8192 tiles total
  const int mtile = tile >> 4;              // 0..511  (row tiles over B*Q)
  const int ntile = tile & 15;              // 0..15   (col tiles over D)
  const int col = lane & 15, half = lane >> 4;

  const int bq = mtile * 16 + (lane & 15);  // this lane's A row
  const int b  = bq >> 9;
  const int q  = bq & 511;

  v8f c = {};
#pragma unroll
  for (int k0 = 0; k0 < DD; k0 += 32) {
    const int hh = k0 >> 5;  // k-chunk == head block (H*HD layout)
    const float* Abase = heads + ((size_t)(hh * BB + b) * QQ + q) * HDIM;
    v16h a;
#pragma unroll
    for (int v = 0; v < 8; ++v) {
      const int kb = ((v >> 2) << 4) | (half << 3) | ((v & 3) << 1);
      a[2 * v]     = (_Float16)Abase[kb];
      a[2 * v + 1] = (_Float16)Abase[kb + 1];
    }
    v16h bm = ldB_rows_f32(Wout + (size_t)k0 * DD + ntile * 16, DD, lane);
    c = wmma_fma(a, bm, c);
  }
#pragma unroll
  for (int r = 0; r < 8; ++r)
    out[(size_t)(mtile * 16 + half * 8 + r) * DD + ntile * 16 + col] = c[r];
}

// ---------------------------------------------------------------------------
extern "C" void kernel_launch(void* const* d_in, const int* in_sizes, int n_in,
                              void* d_out, int out_size, void* d_ws, size_t ws_size,
                              hipStream_t stream) {
  (void)in_sizes; (void)n_in; (void)out_size; (void)ws_size;
  const float* h_fea      = (const float*)d_in[0];
  const float* aux        = (const float*)d_in[1];
  const float* Wq_custom  = (const float*)d_in[2];
  const float* Wq_custom1 = (const float*)d_in[3];
  const float* Wk_custom  = (const float*)d_in[4];
  const float* Wv_custom  = (const float*)d_in[5];
  const float* Wq_charge1 = (const float*)d_in[6];
  const float* Wk_charge  = (const float*)d_in[7];
  const float* Wv_charge  = (const float*)d_in[8];
  const float* W1         = (const float*)d_in[9];
  const float* b1         = (const float*)d_in[10];
  const float* W2         = (const float*)d_in[11];
  const float* b2         = (const float*)d_in[12];
  const float* W_out      = (const float*)d_in[13];

  // workspace layout (f16 projections + f32 heads), ~24.6 MB total
  char* ws = (char*)d_ws;
  const size_t szTask = (size_t)HH * BB * NT * HDIM * sizeof(_Float16);
  const size_t szStat = (size_t)HH * BB * NS * HDIM * sizeof(_Float16);
  _Float16* Qt1 = (_Float16*)ws; ws += szTask;
  _Float16* Qt  = (_Float16*)ws; ws += szTask;
  _Float16* Kc  = (_Float16*)ws; ws += szTask;
  _Float16* Vc  = (_Float16*)ws; ws += szTask;
  _Float16* Qs  = (_Float16*)ws; ws += szStat;
  _Float16* Ks  = (_Float16*)ws; ws += szStat;
  _Float16* Vs  = (_Float16*)ws; ws += szStat;
  float*    heads = (float*)ws;  // HH*BB*QQ*HDIM f32 = 8 MB

  const int projBlocks = 4 * HH * BB * 28 + 3 * HH * BB * 4;  // 15872
  proj_kernel<<<projBlocks, 32, 0, stream>>>(
      h_fea, Wq_custom1, Wq_custom, Wk_custom, Wv_custom,
      Wq_charge1, Wk_charge, Wv_charge,
      Qt1, Qt, Kc, Vc, Qs, Ks, Vs);

  const size_t shT = (size_t)(8 * 16 * (NT + NS) + 408) * sizeof(float);  // 263,776 B
  attn_kernel<true><<<BB * 28, 256, shT, stream>>>(
      Qt1, Qt, Kc, Vc, Qs, Ks, Vs, aux, W1, b1, W2, b2, heads);

  const size_t shS = (size_t)(8 * 16 * NT + 408) * sizeof(float);  // 231,008 B
  attn_kernel<false><<<BB * 4, 256, shS, stream>>>(
      Qt1, Qt, Kc, Vc, Qs, Ks, Vs, aux, W1, b1, W2, b2, heads);

  outproj_kernel<<<1024, 256, 0, stream>>>(heads, W_out, (float*)d_out);
}
